// basicAttn_71064528880123
// MI455X (gfx1250) — matrix operand order repaired and checked by reading, not compiled, not run on hardware
//
#include <hip/hip_runtime.h>

// Problem constants
#define BB 64
#define EE 256
#define HH 256
#define LL 2048
#define G3 768          // 3*H
#define NLBL 5
#define GSTR 772        // padded LDS row stride for g (772%64=4 -> conflict-free scatter)

typedef __attribute__((ext_vector_type(16))) __bf16        v16bf;
typedef __attribute__((ext_vector_type(8)))  float         v8f;
typedef __attribute__((ext_vector_type(4)))  unsigned int  v4u;
typedef __attribute__((ext_vector_type(2)))  unsigned int  v2u;
typedef __attribute__((ext_vector_type(4)))  float         v4f;

union FragAB { v16bf v; v4u q[2]; };

__device__ __forceinline__ unsigned short f2bf(float f) {
  unsigned int u = __float_as_uint(f);
  u += 0x7FFFu + ((u >> 16) & 1u);          // round-to-nearest-even
  return (unsigned short)(u >> 16);
}

// A fragment: 16x32 bf16 tile, row-major source with leading dim ld.
// Lane l: m = row0 + (l&15); elements 0..7 -> K = k0 + (l>>4)*8 + 0..7,
//         elements 8..15 -> K = k0 + 16 + (l>>4)*8 + 0..7  (ISA 7.12.2)
__device__ __forceinline__ FragAB load_a16(const unsigned short* base, int ld,
                                           int row0, int k0, int lane) {
  const unsigned short* p = base + (size_t)(row0 + (lane & 15)) * ld + k0 + ((lane >> 4) << 3);
  FragAB f;
  f.q[0] = *(const v4u*)(p);
  f.q[1] = *(const v4u*)(p + 16);
  return f;
}

// B fragment: 32x16 bf16 tile of W^T, i.e. B[k][n] = W[n][k], W row-major [N x ld].
// Lane l: n = n0 + (l&15); elements 0..7 -> K = k0 + (l>>4)*8 + 0..7,
//         elements 8..15 -> K = k0 + 16 + (l>>4)*8 + 0..7
__device__ __forceinline__ FragAB load_b16(const unsigned short* base, int ld,
                                           int n0, int k0, int lane) {
  const unsigned short* p = base + (size_t)(n0 + (lane & 15)) * ld + k0 + ((lane >> 4) << 3);
  FragAB f;
  f.q[0] = *(const v4u*)(p);
  f.q[1] = *(const v4u*)(p + 16);
  return f;
}

__device__ __forceinline__ v8f wmma_bf16(FragAB a, FragAB b, v8f c) {
  return __builtin_amdgcn_wmma_f32_16x16x32_bf16(false, a.v, false, b.v, (short)0, c, false, false);
}

__device__ __forceinline__ v8f v8zero() {
  v8f z = {0.f,0.f,0.f,0.f,0.f,0.f,0.f,0.f};
  return z;
}

// ---------------- fp32 -> bf16 conversion ----------------
__global__ __launch_bounds__(256) void k_f32_to_bf16(const float* __restrict__ src,
                                                     unsigned short* __restrict__ dst, int n) {
  int i = blockIdx.x * 256 + threadIdx.x;
  if (i < n) dst[i] = f2bf(src[i]);
}

// ---------------- embedding gather -> x_bf16[L*B, E] ----------------
__global__ __launch_bounds__(256) void k_embed(const int* __restrict__ ids,
                                               const float* __restrict__ emb,
                                               unsigned short* __restrict__ x) {
  int idx = blockIdx.x * 256 + threadIdx.x;        // over L*B*E
  int e   = idx & (EE - 1);
  int row = idx >> 8;                              // row = l*B + b
  int b   = row & (BB - 1);
  int l   = row >> 6;
  int tok = ids[b * LL + l];
  x[idx] = f2bf(emb[(size_t)tok * EE + e]);
}

// ---------------- input projection: xW = x @ W_ih^T + b_ih ----------------
// x: [131072 x 256] bf16, w: [768 x 256] bf16, out: [131072 x 768] f32
// block = 256 (8 waves); wave: 2 M-tiles x 4 N-tiles; grid (512, 12)
__global__ __launch_bounds__(256) void k_gemm_inproj(const unsigned short* __restrict__ x,
                                                     const unsigned short* __restrict__ w,
                                                     const float* __restrict__ bias,
                                                     float* __restrict__ out) {
  const int lane = threadIdx.x & 31;
  const int wave = threadIdx.x >> 5;
  const int mt0  = blockIdx.x * 16 + wave * 2;     // first of 2 m-tiles
  const int n0   = blockIdx.y * 64;

  v8f acc[2][4];
  for (int r = 0; r < 2; ++r)
    for (int i = 0; i < 4; ++i) acc[r][i] = v8zero();

  for (int kt = 0; kt < 8; ++kt) {
    FragAB a0 = load_a16(x, EE, mt0 * 16,      kt * 32, lane);
    FragAB a1 = load_a16(x, EE, mt0 * 16 + 16, kt * 32, lane);
    for (int i = 0; i < 4; ++i) {
      FragAB bf = load_b16(w, EE, n0 + i * 16, kt * 32, lane);
      acc[0][i] = wmma_bf16(a0, bf, acc[0][i]);
      acc[1][i] = wmma_bf16(a1, bf, acc[1][i]);
    }
  }
  for (int r = 0; r < 2; ++r) {
    int mbase = (mt0 + r) * 16 + (lane >> 4) * 8;
    for (int i = 0; i < 4; ++i) {
      int col  = n0 + i * 16 + (lane & 15);
      float bv = bias[col];
      for (int v = 0; v < 8; ++v)
        out[(size_t)(mbase + v) * G3 + col] = acc[r][i][v] + bv;
    }
  }
}

// ---------------- persistent GRU scan (1 workgroup per direction) ----------------
// LDS: g[64][GSTR] f32, h_f32[64][256], h_bf16[64][256], b_hh cache[768]
#define SMEM_G   (64 * GSTR * 4)
#define SMEM_HF  (64 * 256 * 4)
#define SMEM_HB  (64 * 256 * 2)
#define SMEM_BHH (G3 * 4)
#define SMEM_BYTES (SMEM_G + SMEM_HF + SMEM_HB + SMEM_BHH)
__global__ __launch_bounds__(1024) void k_gru_scan(const float* __restrict__ xW_f,
                                                   const float* __restrict__ xW_b,
                                                   const unsigned short* __restrict__ whh_f,
                                                   const unsigned short* __restrict__ whh_b,
                                                   const float* __restrict__ bhh_f,
                                                   const float* __restrict__ bhh_b,
                                                   const float* __restrict__ hidden,
                                                   float* __restrict__ hs_f,
                                                   float* __restrict__ hs_b) {
  extern __shared__ char smem[];
  float*          g   = (float*)smem;                                   // 64 x GSTR
  float*          hf  = (float*)(smem + SMEM_G);                        // 64 x 256
  unsigned short* hb  = (unsigned short*)(smem + SMEM_G + SMEM_HF);     // 64 x 256 bf16
  float*          bhs = (float*)(smem + SMEM_G + SMEM_HF + SMEM_HB);    // 768

  const int dir = blockIdx.x;
  const float* xW            = dir ? xW_b  : xW_f;
  const unsigned short* W    = dir ? whh_b : whh_f;
  const float* bhh           = dir ? bhh_b : bhh_f;
  float* hs                  = dir ? hs_b  : hs_f;

  const int tid  = threadIdx.x;
  const int lane = tid & 31;
  const int wave = tid >> 5;
  const int mt   = wave >> 3;        // 4 m-tiles (B=64)
  const int wc   = wave & 7;         // 8 wave columns x 6 n-tiles = 48 n-tiles

  // init h from hidden[dir]; cache b_hh in LDS
  for (int k = tid; k < BB * HH; k += 1024) {
    float v = hidden[dir * BB * HH + k];
    hf[k] = v;
    hb[k] = f2bf(v);
  }
  if (tid < G3) bhs[tid] = bhh[tid];
  __syncthreads();

  for (int s = 0; s < LL; ++s) {
    const int t = dir ? (LL - 1 - s) : s;

    // prefetch next step's gate inputs (768 KB/dir/step) behind the WMMA phase
    if (s + 1 < LL) {
      const int tn = dir ? (LL - 2 - s) : (s + 1);
      const char* np = (const char*)(xW + (size_t)tn * BB * G3);
      #pragma unroll
      for (int j = 0; j < 6; ++j)
        __builtin_prefetch(np + (size_t)tid * 768 + j * 128, 0, 1);
    }

    // ---- g = h @ W_hh^T  (64x256 * 256x768), WMMA bf16, fp32 acc ----
    v8f acc[6];
    for (int i = 0; i < 6; ++i) acc[i] = v8zero();
    for (int kt = 0; kt < 8; ++kt) {
      FragAB a = load_a16(hb, HH, mt * 16, kt * 32, lane);
      for (int i = 0; i < 6; ++i) {
        FragAB bf = load_b16(W, HH, (wc * 6 + i) * 16, kt * 32, lane);
        acc[i] = wmma_bf16(a, bf, acc[i]);
      }
    }
    {
      const int mrow = mt * 16 + (lane >> 4) * 8;
      for (int i = 0; i < 6; ++i) {
        const int col = (wc * 6 + i) * 16 + (lane & 15);
        for (int v = 0; v < 8; ++v)
          g[(mrow + v) * GSTR + col] = acc[i][v];
      }
    }
    __syncthreads();

    // ---- fused GRU gates, fully vectorized (v4f / b128) ----
    const float* xrow = xW + (size_t)t * BB * G3;
    #pragma unroll
    for (int it = 0; it < 4; ++it) {
      const int k4 = (tid + it * 1024) * 4;        // 4 consecutive elements
      const int b = k4 >> 8, c = k4 & 255;
      const float* xp = xrow + b * G3;
      v4f xr = *(const v4f*)(xp + c);
      v4f xz = *(const v4f*)(xp + HH + c);
      v4f xn = *(const v4f*)(xp + 2 * HH + c);
      v4f gr = *(const v4f*)(&g[b * GSTR + c]);
      v4f gz = *(const v4f*)(&g[b * GSTR + HH + c]);
      v4f gn = *(const v4f*)(&g[b * GSTR + 2 * HH + c]);
      v4f br = *(const v4f*)(&bhs[c]);
      v4f bz = *(const v4f*)(&bhs[HH + c]);
      v4f bn = *(const v4f*)(&bhs[2 * HH + c]);
      v4f hold = *(const v4f*)(&hf[k4]);
      v4f hnew;
      #pragma unroll
      for (int j = 0; j < 4; ++j) {
        float r = 1.f / (1.f + __expf(-(xr[j] + gr[j] + br[j])));
        float z = 1.f / (1.f + __expf(-(xz[j] + gz[j] + bz[j])));
        float n = tanhf(xn[j] + r * (gn[j] + bn[j]));
        hnew[j] = (1.f - z) * n + z * hold[j];
      }
      *(v4f*)(&hf[k4]) = hnew;
      v2u pk = { (unsigned)f2bf(hnew[0]) | ((unsigned)f2bf(hnew[1]) << 16),
                 (unsigned)f2bf(hnew[2]) | ((unsigned)f2bf(hnew[3]) << 16) };
      *(v2u*)(&hb[k4]) = pk;
      *(v4f*)(&hs[((size_t)t * BB + b) * HH + c]) = hnew;
    }
    __syncthreads();
  }
}

// ---------------- attention scores: e[b][l] = tanh(rnn . aw + ab), wave per output ----------------
__global__ __launch_bounds__(256) void k_escore(const float* __restrict__ hs_f,
                                                const float* __restrict__ hs_b,
                                                const float* __restrict__ aw,
                                                const float* __restrict__ ab,
                                                float* __restrict__ e) {
  const int wave = threadIdx.x >> 5, lane = threadIdx.x & 31;
  const int o = blockIdx.x * 8 + wave;     // [0, B*L)
  const int b = o >> 11, l = o & (LL - 1);
  const float* src = (lane < 16)
      ? (hs_f + ((size_t)l * BB + b) * HH + lane * 16)
      : (hs_b + ((size_t)l * BB + b) * HH + (lane - 16) * 16);
  const float* wp = aw + lane * 16;        // aw[0:256]=fwd, aw[256:512]=bwd
  float s = 0.f;
  for (int i = 0; i < 16; i += 4) {
    v4f hv = *(const v4f*)(src + i);
    v4f wv = *(const v4f*)(wp + i);
    s += hv.x * wv.x + hv.y * wv.y + hv.z * wv.z + hv.w * wv.w;
  }
  for (int m = 16; m >= 1; m >>= 1) s += __shfl_xor(s, m, 32);
  if (lane == 0) e[o] = tanhf(s + ab[0]);
}

// ---------------- softmax over L per batch row ----------------
__global__ __launch_bounds__(256) void k_softmax(const float* __restrict__ e,
                                                 float* __restrict__ alpha) {
  __shared__ float red[256];
  const int b = blockIdx.x, tid = threadIdx.x;
  const float* eb = e + b * LL;
  float m = -1e30f;
  for (int l = tid; l < LL; l += 256) m = fmaxf(m, eb[l]);
  red[tid] = m; __syncthreads();
  for (int s2 = 128; s2 > 0; s2 >>= 1) {
    if (tid < s2) red[tid] = fmaxf(red[tid], red[tid + s2]);
    __syncthreads();
  }
  const float bmax = red[0];
  __syncthreads();
  float sum = 0.f;
  for (int l = tid; l < LL; l += 256) {
    float v = __expf(eb[l] - bmax);
    alpha[b * LL + l] = v;
    sum += v;
  }
  red[tid] = sum; __syncthreads();
  for (int s2 = 128; s2 > 0; s2 >>= 1) {
    if (tid < s2) red[tid] += red[tid + s2];
    __syncthreads();
  }
  const float inv = 1.f / red[0];
  for (int l = tid; l < LL; l += 256) alpha[b * LL + l] *= inv;
}

// ---------------- context[b][d] = sum_l alpha[b][l] * rnn[l][b][d] ----------------
__global__ __launch_bounds__(256) void k_context(const float* __restrict__ hs_f,
                                                 const float* __restrict__ hs_b,
                                                 const float* __restrict__ alpha,
                                                 float* __restrict__ ctx) {
  const int b = blockIdx.x >> 1, half = blockIdx.x & 1;
  const int d = half * 256 + threadIdx.x;          // 0..511
  const float* hs = (d < HH) ? hs_f : hs_b;
  const int dd = d & (HH - 1);
  float acc = 0.f;
  for (int l = 0; l < LL; ++l)
    acc += alpha[b * LL + l] * hs[((size_t)l * BB + b) * HH + dd];
  ctx[b * (2 * HH) + d] = acc;
}

// ---------------- logits + hidden_out ----------------
__global__ __launch_bounds__(256) void k_final(const float* __restrict__ ctx,
                                               const float* __restrict__ out_w,
                                               const float* __restrict__ out_b,
                                               const float* __restrict__ hs_f,
                                               const float* __restrict__ hs_b,
                                               float* __restrict__ out) {
  const int idx = blockIdx.x * 256 + threadIdx.x;
  if (idx < BB * NLBL) {
    const int b = idx / NLBL, lab = idx % NLBL;
    float s = out_b[lab];
    for (int d = 0; d < 2 * HH; ++d)
      s += ctx[b * (2 * HH) + d] * out_w[lab * (2 * HH) + d];
    out[idx] = s;                                          // logits [64,5]
  } else if (idx < BB * NLBL + 2 * BB * HH) {
    const int j = idx - BB * NLBL;                         // hidden [2,64,256]
    const int d = j & 255, b = (j >> 8) & 63, dirn = j >> 14;
    out[idx] = (dirn == 0)
        ? hs_f[((size_t)(LL - 1) * BB + b) * HH + d]       // hlast_f = hs_f[L-1]
        : hs_b[(size_t)b * HH + d];                        // hlast_b = hs_b[0]
  }
}

extern "C" void kernel_launch(void* const* d_in, const int* in_sizes, int n_in,
                              void* d_out, int out_size, void* d_ws, size_t ws_size,
                              hipStream_t stream) {
  (void)in_sizes; (void)n_in; (void)out_size; (void)ws_size;
  const int*   ids    = (const int*)d_in[0];
  const float* hidden = (const float*)d_in[2];
  const float* emb    = (const float*)d_in[3];
  const float* Wihf   = (const float*)d_in[4];
  const float* Whhf   = (const float*)d_in[5];
  const float* bihf   = (const float*)d_in[6];
  const float* bhhf   = (const float*)d_in[7];
  const float* Wihb   = (const float*)d_in[8];
  const float* Whhb   = (const float*)d_in[9];
  const float* bihb   = (const float*)d_in[10];
  const float* bhhb   = (const float*)d_in[11];
  const float* aw     = (const float*)d_in[12];
  const float* ab     = (const float*)d_in[13];
  const float* ow     = (const float*)d_in[14];
  const float* ob     = (const float*)d_in[15];
  float* out = (float*)d_out;

  // workspace layout (bytes, all 16B aligned)
  char* ws = (char*)d_ws;
  unsigned short* wihf_bf = (unsigned short*)(ws + 0);
  unsigned short* wihb_bf = (unsigned short*)(ws + 393216ULL);
  unsigned short* whhf_bf = (unsigned short*)(ws + 786432ULL);
  unsigned short* whhb_bf = (unsigned short*)(ws + 1179648ULL);
  unsigned short* x_bf    = (unsigned short*)(ws + 1572864ULL);     // 64 MB
  float* xWf = (float*)(ws + 68681728ULL);                          // 402 MB
  float* xWb = (float*)(ws + 471334912ULL);                         // 402 MB
  float* hsf = (float*)(ws + 873988096ULL);                         // 128 MB
  float* hsb = (float*)(ws + 1008205824ULL);                        // 128 MB
  float* e_s = (float*)(ws + 1142423552ULL);                        // B*L
  float* ctx = (float*)(ws + 1142947840ULL);                        // B*512

  const int NW = G3 * EE;  // 196608 elements per weight matrix
  k_f32_to_bf16<<<NW / 256, 256, 0, stream>>>(Wihf, wihf_bf, NW);
  k_f32_to_bf16<<<NW / 256, 256, 0, stream>>>(Wihb, wihb_bf, NW);
  k_f32_to_bf16<<<NW / 256, 256, 0, stream>>>(Whhf, whhf_bf, NW);
  k_f32_to_bf16<<<NW / 256, 256, 0, stream>>>(Whhb, whhb_bf, NW);

  k_embed<<<(LL * BB * EE) / 256, 256, 0, stream>>>(ids, emb, x_bf);

  dim3 gg(512, 12, 1);
  k_gemm_inproj<<<gg, 256, 0, stream>>>(x_bf, wihf_bf, bihf, xWf);
  k_gemm_inproj<<<gg, 256, 0, stream>>>(x_bf, wihb_bf, bihb, xWb);

  (void)hipFuncSetAttribute((const void*)k_gru_scan,
                            hipFuncAttributeMaxDynamicSharedMemorySize, SMEM_BYTES);
  k_gru_scan<<<2, 1024, SMEM_BYTES, stream>>>(xWf, xWb, whhf_bf, whhb_bf,
                                              bhhf, bhhb, hidden, hsf, hsb);

  float* alpha = out + BB * NLBL + 2 * BB * HH;   // alpha region of d_out
  k_escore<<<(BB * LL) / 8, 256, 0, stream>>>(hsf, hsb, aw, ab, e_s);
  k_softmax<<<BB, 256, 0, stream>>>(e_s, alpha);
  k_context<<<2 * BB, 256, 0, stream>>>(hsf, hsb, alpha, ctx);
  k_final<<<(BB * NLBL + 2 * BB * HH + 255) / 256, 256, 0, stream>>>(ctx, ow, ob, hsf, hsb, out);
}
